// SubmanifoldConvolution_13469017440654
// MI455X (gfx1250) — matrix operand order repaired and checked by reading, not compiled, not run on hardware
//
#include <hip/hip_runtime.h>
#include <hip/hip_bf16.h>

typedef __attribute__((ext_vector_type(16))) __bf16 v16bf;
typedef __attribute__((ext_vector_type(8)))  float  v8f;

#define N_ACTIVE 200000
#define N_IN  64
#define N_OUT 64
#define KF    27
#define RB    100000
#define ROWS_PER_BLOCK 1024                       // 8 waves * 8 iters * 16 rows
#define CHUNKS ((RB + ROWS_PER_BLOCK - 1) / ROWS_PER_BLOCK)   // 98

// out[i][j] = bias[j]  (also clears the poisoned output buffer)
__global__ __launch_bounds__(256) void init_out_kernel(const float* __restrict__ bias,
                                                       float* __restrict__ out, int total) {
  int i = blockIdx.x * 256 + threadIdx.x;
  if (i < total) out[i] = bias[i & (N_OUT - 1)];
}

__device__ inline void cvt8(float4 a, float4 b, v16bf& d, int off) {
  d[off + 0] = (__bf16)a.x; d[off + 1] = (__bf16)a.y;
  d[off + 2] = (__bf16)a.z; d[off + 3] = (__bf16)a.w;
  d[off + 4] = (__bf16)b.x; d[off + 5] = (__bf16)b.y;
  d[off + 6] = (__bf16)b.z; d[off + 7] = (__bf16)b.w;
}

__global__ __launch_bounds__(256) void spconv_wmma_kernel(
    const float* __restrict__ features,   // [N_ACTIVE, 64] f32
    const float* __restrict__ weight,     // [27, 64, 64]  f32 (k, nIn, nOut)
    const int*   __restrict__ in_idx,     // [27, R]
    const int*   __restrict__ out_idx,    // [27, R]
    float*       __restrict__ out)        // [N_ACTIVE, 64] f32 (pre-initialized with bias)
{
  // 8 B tiles (nt=0..3, kt=0..1), each [32 lanes][16 bf16] = 32B/lane, 8 KB total,
  // stored pre-swizzled in the WMMA B-operand lane layout.
  __shared__ uint4 Bs[8 * 32 * 2];

  const int k     = blockIdx.y;
  const int chunk = blockIdx.x;
  const int tid   = threadIdx.x;

  // ---- stage this k's 64x64 weight slice into LDS, bf16, B-operand layout ----
  {
    const int tile = tid >> 5;          // 0..7 : tile = nt*2 + kt
    const int lane = tid & 31;
    const int shi  = lane >> 4;
    const int snl  = lane & 15;
    const int nt   = tile >> 1;
    const int kt   = tile & 1;
    const int n    = nt * 16 + snl;                    // output column
    const int ks   = kt * 32 + shi * 8;                // first K of this lane's first 8 (then +16)
    const float* wk = weight + k * (N_IN * N_OUT);
    __bf16 tmp[16] __attribute__((aligned(16)));
    #pragma unroll
    for (int j = 0; j < 16; ++j)
      tmp[j] = (__bf16)wk[(ks + j + (j >= 8 ? 8 : 0)) * N_OUT + n];       // B[K][N], pairs (2j,2j+1)/VGPR
    uint4* dst = &Bs[(tile * 32 + lane) * 2];
    dst[0] = ((const uint4*)tmp)[0];
    dst[1] = ((const uint4*)tmp)[1];
  }
  __syncthreads();

  const int wave = tid >> 5;
  const int lane = tid & 31;
  const int hi   = lane >> 4;
  const int nl   = lane & 15;

  // ---- pull all 8 B tiles into VGPRs (reused for all row-tiles of this block) ----
  v16bf Breg[8];
  #pragma unroll
  for (int t = 0; t < 8; ++t)
    Breg[t] = *(const v16bf*)&Bs[(t * 32 + lane) * 2];  // 2x ds_load_b128 each

  const int* in_k  = in_idx  + k * RB;
  const int* out_k = out_idx + k * RB;

  const int wbase = chunk * ROWS_PER_BLOCK + wave * 128;

  // software pipeline: preload the first gather row index
  int r0   = wbase + nl;
  int grow = in_k[r0 < RB ? r0 : (RB - 1)];

  #pragma unroll 1
  for (int it = 0; it < 8; ++it) {
    const int rbase = wbase + it * 16;

    // ---- gather one 16x64 feature tile; lane pair (L, L+16) shares row M = L&15 ----
    const float4* fr = (const float4*)(features + (long)grow * N_IN);

    // prefetch next iteration's feature row (random gather -> warm WGP cache)
    int grow_next = grow;
    if (it < 7) {
      int rn = rbase + 16 + nl;
      grow_next = in_k[rn < RB ? rn : (RB - 1)];
      const char* nf = (const char*)(features + (long)grow_next * N_IN);
      __builtin_prefetch(nf, 0, 3);          // global_prefetch_b8, first 128B
      __builtin_prefetch(nf + 128, 0, 3);    // second half of the 256B row
    }

    // A layout (16-bit A 16x32): hi=0 -> K{0..7,16..23}; hi=1 -> K{8..15,24..31}
    float4 p0 = fr[hi * 2 + 0],  p1 = fr[hi * 2 + 1];       // K  0..7  (+hi*8)
    float4 p2 = fr[4 + hi * 2],  p3 = fr[5 + hi * 2];       // K 16..23 (+hi*8)
    float4 p4 = fr[8 + hi * 2],  p5 = fr[9 + hi * 2];       // K 32..39 (+hi*8)
    float4 p6 = fr[12 + hi * 2], p7 = fr[13 + hi * 2];      // K 48..55 (+hi*8)
    v16bf a0, a1;
    cvt8(p0, p1, a0, 0); cvt8(p2, p3, a0, 8);
    cvt8(p4, p5, a1, 0); cvt8(p6, p7, a1, 8);

    // ---- scatter rows for this lane: M = hi*8 + g ----
    int  orow[8]; bool ovalid[8];
    #pragma unroll
    for (int g = 0; g < 8; ++g) {
      int rr    = rbase + hi * 8 + g;
      ovalid[g] = rr < RB;
      orow[g]   = ovalid[g] ? out_k[rr] : 0;
    }

    #pragma unroll
    for (int nt = 0; nt < 4; ++nt) {
      v8f acc = {};
      acc = __builtin_amdgcn_wmma_f32_16x16x32_bf16(false, a0, false, Breg[nt * 2 + 0],
                                                    (short)0, acc, false, false);
      acc = __builtin_amdgcn_wmma_f32_16x16x32_bf16(false, a1, false, Breg[nt * 2 + 1],
                                                    (short)0, acc, false, false);
      const int col = nt * 16 + nl;
      #pragma unroll
      for (int g = 0; g < 8; ++g) {
        if (ovalid[g]) {
          // native non-returning global_atomic_add_f32 (no CAS fallback)
          (void)__hip_atomic_fetch_add(out + orow[g] * N_OUT + col, acc[g],
                                       __ATOMIC_RELAXED, __HIP_MEMORY_SCOPE_AGENT);
        }
      }
    }

    grow = grow_next;
  }
}

extern "C" void kernel_launch(void* const* d_in, const int* in_sizes, int n_in_args,
                              void* d_out, int out_size, void* d_ws, size_t ws_size,
                              hipStream_t stream) {
  const float* features = (const float*)d_in[0];
  const float* weight   = (const float*)d_in[1];
  const float* bias     = (const float*)d_in[2];
  const int*   in_idx   = (const int*)d_in[3];
  const int*   out_idx  = (const int*)d_in[4];
  float* out = (float*)d_out;

  const int total = N_ACTIVE * N_OUT;
  init_out_kernel<<<(total + 255) / 256, 256, 0, stream>>>(bias, out, total);

  dim3 grid(CHUNKS, KF);   // 98 x 27 blocks, 256 threads (8 wave32) each
  spconv_wmma_kernel<<<grid, 256, 0, stream>>>(features, weight, in_idx, out_idx, out);
}
